// Distribution_70729521430732
// MI455X (gfx1250) — compile-verified, matrix operand order checked by reading, not run on hardware
//
#include <hip/hip_runtime.h>
#include <hip/hip_bf16.h>

// ---------------------------------------------------------------------------
// Linear-chain CRF log-partition via chunked log-semiring scan.
//
// Roofline: input = 64*256*128*128*4 B = 1.07 GB read once -> 46 us floor at
// 23.3 TB/s. Sequential DP exposes only 64 WGs; instead we compute 16-segment
// log-semiring matrix products (64*16 = 1024 independent WGs) where each
// product is a stability-shifted exp-domain GEMM executed on
// v_wmma_f32_16x16x32_f16 (shifted exp values are in (0,1] -> f16-safe,
// f32 accumulate). Total WMMA work ~64 GFLOP f16 -> negligible; the kernel
// stays HBM-bound. Tiles are streamed global->LDS by the Tensor Data Mover
// (tensor_load_to_lds, double-buffered, TENSORcnt-synced) with TDM row
// padding (129-float row stride) to remove LDS bank conflicts.
// ---------------------------------------------------------------------------

#define LOG2E 1.44269504088896340736f
#define LN2   0.69314718055994530942f

typedef __attribute__((ext_vector_type(16))) _Float16 v16h;
typedef __attribute__((ext_vector_type(8)))  _Float16 v8h;
typedef __attribute__((ext_vector_type(8)))  float    v8f;
typedef __attribute__((ext_vector_type(4)))  unsigned u32x4;
typedef __attribute__((ext_vector_type(8)))  int      i32x8;
typedef __attribute__((ext_vector_type(4)))  int      i32x4;

union F16Frag { v16h v; v8h h[2]; };

static constexpr int T        = 128;        // label count
static constexpr int FSTRIDE  = 129;        // f32 LDS row stride (odd -> no bank conflicts)
static constexpr int HSTRIDE  = 136;        // f16 LDS row stride (16B-aligned rows, staggered banks)
static constexpr unsigned OFF_BUF0 = 0;                   // streamed tile, f32 128x129
static constexpr unsigned OFF_BUF1 = 66048;               // 128*129*4
static constexpr unsigned OFF_P    = 132096;              // running product, f32 128x129
static constexpr unsigned OFF_A16  = 198144;              // exp(P - r_i), f16 128x136
static constexpr unsigned OFF_B16  = 232960;              // exp(M - s_j) transposed, f16 128x136
static constexpr unsigned OFF_R    = 267776;              // row shifts r_i (128 f32)
static constexpr unsigned OFF_S    = 268288;              // col shifts s_j (128 f32)
static constexpr unsigned SMEM_BYTES = 268800;            // < 320 KB WGP LDS

__device__ __forceinline__ float fexp(float x) { return __builtin_amdgcn_exp2f(x * LOG2E); }
__device__ __forceinline__ float flog(float x) { return __builtin_amdgcn_logf(x) * LN2; }

// Issue one TDM copy: 128x128 f32 tile, global -> LDS, with 1-DWORD pad after
// every 128 DWORDs so LDS rows land at a 129-float stride.
__device__ __forceinline__ void tdm_load_tile(const float* gptr, unsigned lds_off) {
  unsigned long long ga = (unsigned long long)(uintptr_t)gptr;
  u32x4 g0;
  g0[0] = 1u;                                        // count=1, user descriptor
  g0[1] = lds_off;                                   // LDS byte address
  g0[2] = (unsigned)(ga & 0xFFFFFFFFu);              // global_addr[31:0]
  g0[3] = (unsigned)((ga >> 32) & 0x1FFFFFFu)        // global_addr[56:32]
          | 0x80000000u;                             // type=2 ("image")
  i32x8 g1;
  g1[0] = (int)((2u << 16)      // data_size = 4B
              | (1u << 20)      // pad_enable
              | (6u << 22));    // pad_interval = 128 DWORDs; pad_amount=0 -> 1 DWORD
  g1[1] = (int)(128u << 16);    // tensor_dim0[15:0] = 128
  g1[2] = (int)(128u << 16);    // tensor_dim0 hi = 0 | tensor_dim1[15:0] = 128
  g1[3] = (int)(128u << 16);    // tensor_dim1 hi = 0 | tile_dim0 = 128
  g1[4] = (int)128;             // tile_dim1 = 128, tile_dim2 = 0 (unused)
  g1[5] = 128;                  // tensor_dim0_stride = 128 elements
  g1[6] = 0;
  g1[7] = 0;
  i32x4 gz4 = {0, 0, 0, 0};                  // 2D tensor: groups 2/3 unused
  i32x8 gz8 = {0, 0, 0, 0, 0, 0, 0, 0};      // trailing group (6-arg form): unused
  __builtin_amdgcn_tensor_load_to_lds(g0, g1, gz4, gz4, gz8, 0);
}

// ---------------------------------------------------------------------------
// Phase 1: WG (b,s) computes P = M[16s] (x) M[16s+1] (x) ... (x) M[16s+15]
// in the log semiring. 256 threads = 8 waves.
// ---------------------------------------------------------------------------
__global__ __launch_bounds__(256) void crf_segprod_kernel(const float* __restrict__ lp,
                                                          float* __restrict__ segout) {
  extern __shared__ char smem[];
  float* buf0 = (float*)(smem + OFF_BUF0);
  float* buf1 = (float*)(smem + OFF_BUF1);
  float* P    = (float*)(smem + OFF_P);
  float* rbuf = (float*)(smem + OFF_R);
  float* sbuf = (float*)(smem + OFF_S);

  const int tid = threadIdx.x;
  const int b   = blockIdx.x >> 4;
  const int s   = blockIdx.x & 15;
  const long long tilebase = (long long)(b * 256 + s * 16) * (T * T);

  // Prologue: wave 0 issues async TDM loads for tiles 0 and 1.
  if (tid < 32) {
    tdm_load_tile(lp + tilebase, OFF_BUF0);
    tdm_load_tile(lp + tilebase + (long long)T * T, OFF_BUF1);
  }

  for (int t = 1; t < 16; ++t) {
    // Wave 0 owns TENSORcnt; barrier publishes LDS readiness to all waves.
    if (tid < 32) {
      if (t == 1) __builtin_amdgcn_s_wait_tensorcnt(0);  // need tiles 0 AND 1
      else        __builtin_amdgcn_s_wait_tensorcnt(1);  // oldest (tile t) done
    }
    __syncthreads();

    // ---- A panel: A16[i][k] = f16(exp(S[i][k] - r_i)), S = P (or tile0 at t==1)
    {
      const float* S = (t == 1) ? buf0 : P;
      const int row = tid >> 1;
      const int k0  = (tid & 1) * 64;
      const float* sr = S + row * FSTRIDE + k0;
      float mx = -3.4e38f;
      #pragma unroll 8
      for (int k = 0; k < 64; ++k) mx = fmaxf(mx, sr[k]);
      mx = fmaxf(mx, __shfl_xor(mx, 1));               // pair-combine halves
      if ((tid & 1) == 0) rbuf[row] = mx;
      unsigned* dst = (unsigned*)(smem + OFF_A16 + row * (HSTRIDE * 2) + k0 * 2);
      #pragma unroll 8
      for (int k = 0; k < 64; k += 2) {
        _Float16 f0 = (_Float16)fexp(sr[k]     - mx);
        _Float16 f1 = (_Float16)fexp(sr[k + 1] - mx);
        dst[k >> 1] = (unsigned)__builtin_bit_cast(unsigned short, f0)
                    | ((unsigned)__builtin_bit_cast(unsigned short, f1) << 16);
      }
    }
    // ---- B panel (stored transposed): B16[j][k] = f16(exp(M[k][j] - s_j))
    {
      const float* Tt = (t & 1) ? buf1 : buf0;
      const int col = tid >> 1;
      const int k0  = (tid & 1) * 64;
      float mx = -3.4e38f;
      #pragma unroll 8
      for (int k = 0; k < 64; ++k) mx = fmaxf(mx, Tt[(k0 + k) * FSTRIDE + col]);
      mx = fmaxf(mx, __shfl_xor(mx, 1));
      if ((tid & 1) == 0) sbuf[col] = mx;
      unsigned* dst = (unsigned*)(smem + OFF_B16 + col * (HSTRIDE * 2) + k0 * 2);
      #pragma unroll 4
      for (int k = 0; k < 64; k += 2) {
        _Float16 f0 = (_Float16)fexp(Tt[(k0 + k)     * FSTRIDE + col] - mx);
        _Float16 f1 = (_Float16)fexp(Tt[(k0 + k + 1) * FSTRIDE + col] - mx);
        dst[k >> 1] = (unsigned)__builtin_bit_cast(unsigned short, f0)
                    | ((unsigned)__builtin_bit_cast(unsigned short, f1) << 16);
      }
    }
    __syncthreads();

    // Prefetch: both tile buffers are now consumed -> refill while WMMA runs.
    if (tid < 32) {
      if (t == 1) {
        tdm_load_tile(lp + tilebase + 2LL * T * T, OFF_BUF0);
        tdm_load_tile(lp + tilebase + 3LL * T * T, OFF_BUF1);
      } else if (t + 2 < 16) {
        tdm_load_tile(lp + tilebase + (long long)(t + 2) * T * T,
                      (t & 1) ? OFF_BUF1 : OFF_BUF0);
      }
    }

    // ---- exp-domain GEMM on WMMA: C = A16 x B16 ; P = r_i + s_j + ln(C)
    {
      const int wave = tid >> 5, lane = tid & 31;
      const int i0   = wave * 16;          // each wave owns one 16-row stripe
      const int lrow = lane & 15;
      const int sel  = lane >> 4;          // ISA A/B fragment half-select
      F16Frag a[4];
      #pragma unroll
      for (int kc = 0; kc < 4; ++kc) {     // K chunks of 32
        const char* ab = smem + OFF_A16 + (i0 + lrow) * (HSTRIDE * 2) + kc * 64 + sel * 16;
        a[kc].h[0] = *(const v8h*)(ab);         // K = kc*32 + sel*8 .. +7
        a[kc].h[1] = *(const v8h*)(ab + 32);    // K = kc*32 + 16 + sel*8 .. +7
      }
      #pragma unroll 2
      for (int jt = 0; jt < 8; ++jt) {
        v8f c = {0.f, 0.f, 0.f, 0.f, 0.f, 0.f, 0.f, 0.f};
        #pragma unroll
        for (int kc = 0; kc < 4; ++kc) {
          F16Frag bf;
          const char* bb = smem + OFF_B16 + (jt * 16 + lrow) * (HSTRIDE * 2) + kc * 64 + sel * 16;
          bf.h[0] = *(const v8h*)(bb);
          bf.h[1] = *(const v8h*)(bb + 32);
          c = __builtin_amdgcn_wmma_f32_16x16x32_f16(false, a[kc].v, false, bf.v,
                                                     (short)0, c, false, false);
        }
        const int col = jt * 16 + lrow;
        const float sv = sbuf[col];
        const int rbase = i0 + sel * 8;
        #pragma unroll
        for (int r = 0; r < 8; ++r) {
          P[(rbase + r) * FSTRIDE + col] = rbuf[rbase + r] + sv + flog(c[r]);
        }
      }
    }
    __syncthreads();
  }

  // Write unpadded 128x128 segment product to workspace.
  float* o = segout + (long long)blockIdx.x * (T * T);
  for (int idx = tid; idx < T * T; idx += 256)
    o[idx] = P[(idx >> 7) * FSTRIDE + (idx & 127)];
}

// ---------------------------------------------------------------------------
// Phase 2: per batch, fold the 16 segment products with exact two-pass
// logsumexp matvecs, then reduce. 64 WGs x 128 threads; 64 MB traffic total.
// ---------------------------------------------------------------------------
__global__ __launch_bounds__(128) void crf_combine_kernel(const float* __restrict__ segp,
                                                          float* __restrict__ out) {
  __shared__ float va[128];
  __shared__ float red[4];
  const int tid = threadIdx.x;
  const float* seg = segp + (long long)blockIdx.x * 16 * (T * T);

  va[tid] = seg[tid];                      // alpha = row 0 of segment-0 product
  __syncthreads();

  for (int s2 = 1; s2 < 16; ++s2) {
    const float* Ps = seg + s2 * (T * T);
    float tj = -3.4e38f;
    for (int i = 0; i < 128; ++i) tj = fmaxf(tj, va[i] + Ps[i * 128 + tid]);
    float acc = 0.f;
    for (int i = 0; i < 128; ++i) acc += fexp(va[i] + Ps[i * 128 + tid] - tj);
    float vn = tj + flog(acc);
    __syncthreads();
    va[tid] = vn;
    __syncthreads();
  }

  float m = va[tid];
  #pragma unroll
  for (int off = 16; off > 0; off >>= 1) m = fmaxf(m, __shfl_xor(m, off));
  if ((tid & 31) == 0) red[tid >> 5] = m;
  __syncthreads();
  const float mm = fmaxf(fmaxf(red[0], red[1]), fmaxf(red[2], red[3]));
  float e = fexp(va[tid] - mm);
  #pragma unroll
  for (int off = 16; off > 0; off >>= 1) e += __shfl_xor(e, off);
  __syncthreads();
  if ((tid & 31) == 0) red[tid >> 5] = e;
  __syncthreads();
  if (tid == 0) out[blockIdx.x] = mm + flog(red[0] + red[1] + red[2] + red[3]);
}

extern "C" void kernel_launch(void* const* d_in, const int* in_sizes, int n_in,
                              void* d_out, int out_size, void* d_ws, size_t ws_size,
                              hipStream_t stream) {
  (void)in_sizes; (void)n_in; (void)out_size; (void)ws_size;
  const float* lp = (const float*)d_in[0];
  float* segp = (float*)d_ws;              // needs 64*16*16384*4 = 64 MiB scratch
  float* outp = (float*)d_out;

  (void)hipFuncSetAttribute(reinterpret_cast<const void*>(crf_segprod_kernel),
                            hipFuncAttributeMaxDynamicSharedMemorySize, (int)SMEM_BYTES);
  crf_segprod_kernel<<<dim3(1024), dim3(256), SMEM_BYTES, stream>>>(lp, segp);
  crf_combine_kernel<<<dim3(64), dim3(128), 0, stream>>>(segp, outp);
}
